// STGCN2DModel_42468636623278
// MI455X (gfx1250) — matrix operand order repaired and checked by reading, not compiled, not run on hardware
//
#include <hip/hip_runtime.h>
#include <hip/hip_bf16.h>

typedef _Float16 f16;
typedef __attribute__((ext_vector_type(16))) _Float16 v16h;
typedef __attribute__((ext_vector_type(8)))  _Float16 v8h;
typedef __attribute__((ext_vector_type(8)))  float    v8f;

#define DEV __device__ __forceinline__

constexpr int NB   = 512;            // batch
constexpr int TT   = 8;              // time
constexpr int VV   = 96;             // vertices
constexpr int NV   = NB * VV;        // 49152 (LSTM batch)
constexpr int NTV  = NB * TT * VV;   // 393216
constexpr int DYNC = 64;
constexpr int PRED = 12;

DEV float lrelu(float x) { return x > 0.f ? x : 0.1f * x; }
DEV float sigm (float x) { return 1.f / (1.f + __expf(-x)); }

DEV v8f wmma(v16h a, v16h b, v8f c) {
  return __builtin_amdgcn_wmma_f32_16x16x32_f16(false, a, false, b, (short)0, c, false, false);
}

// ---- A fragment (16x32 f16), rows row-major, fast vector path (pitch % 8 == 0)
// lane layout (contiguous k order, same as the B packing): m = lane&15, half = lane>>4;
// a[0..7]  = K = kBase + 16*half + 0..7
// a[8..15] = K = kBase + 16*half + 8 + 0..7
DEV v16h ldA(const f16* base, int pitch, int kBase, int lane) {
  int m = lane & 15, hh = lane >> 4;
  const f16* p = base + (long)m * pitch + kBase + 16 * hh;
  v8h lo = *(const v8h*)p;
  v8h hi = *(const v8h*)(p + 8);
  v16h a;
#pragma unroll
  for (int i = 0; i < 8; ++i) { a[i] = lo[i]; a[i + 8] = hi[i]; }
  return a;
}

// ---- generic scalar A loader: K bound (zero-fill pad) + optional leaky-relu on load
template <bool LR>
DEV v16h ldA_g(const f16* base, int pitch, int kBase, int klim, int lane) {
  int m = lane & 15, hh = lane >> 4;
  v16h a;
#pragma unroll
  for (int i = 0; i < 16; ++i) {
    int k = kBase + 16 * hh + i;
    float x = (k < klim) ? (float)base[(long)m * pitch + k] : 0.f;
    if (LR) x = lrelu(x);
    a[i] = (f16)x;
  }
  return a;
}

// ---- B fragment from fragment-packed weights: frag = 32 lanes x 16 halves contiguous
DEV v16h ldB_pack(const f16* frag, int lane) {
  const f16* q = frag + lane * 16;
  v8h lo = *(const v8h*)q;
  v8h hi = *(const v8h*)(q + 8);
  v16h b;
#pragma unroll
  for (int i = 0; i < 8; ++i) { b[i] = lo[i]; b[i + 8] = hi[i]; }
  return b;
}

// ---- B fragment direct from row-major B[k][n] (pitch = N), used for per-sample adjacency
// lane layout: n = nBase + (lane&15); b[i] = B[kBase + 16*(lane>>4) + i][n]
DEV v16h ldB_dir(const f16* B, int pitch, int kBase, int nBase, int lane) {
  int n  = nBase + (lane & 15);
  int k0 = kBase + ((lane >> 4) << 4);
  v16h b;
#pragma unroll
  for (int i = 0; i < 16; ++i) b[i] = B[(long)(k0 + i) * pitch + n];
  return b;
}

// ======================= small utility kernels =======================

__global__ void k_zero(unsigned int* p, long n) {
  long i = (long)blockIdx.x * blockDim.x + threadIdx.x;
  long stride = (long)gridDim.x * blockDim.x;
  for (; i < n; i += stride) p[i] = 0u;
}

__global__ void k_cvt(const float* __restrict__ s, f16* __restrict__ d, long n) {
  long i = (long)blockIdx.x * blockDim.x + threadIdx.x;
  if (i < n) d[i] = (f16)s[i];
}

__global__ void k_addb(const float* a, const float* b, float* o, int n) {
  int i = blockIdx.x * blockDim.x + threadIdx.x;
  if (i < n) o[i] = a[i] + b[i];
}

// pack W[Nout][Kin] (torch layout, f32) into fragment-packed f16 B operands.
// logical B[k][n] = W[n][k - kbase]; dst frag index = ntile*kctot + kchunk
__global__ void k_pack_w(const float* __restrict__ W, f16* __restrict__ dst,
                         int Nout, int Kin, int kctot, int kbase) {
  int idx = blockIdx.x * 256 + threadIdx.x;
  if (idx >= Nout * Kin) return;
  int n = idx / Kin, k = idx % Kin;
  int kg = k + kbase;
  int nt = n >> 4, nl = n & 15;
  int kc = kg >> 5, kl = kg & 31;
  int lane = ((kl >> 4) << 4) | nl;
  int i = kl & 15;
  dst[(((long)(nt * kctot + kc) * 32 + lane) << 4) + i] = (f16)W[idx];
}

// dyn projection: seq[t][nv][64] = lrelu(x @ Wdyn^T + bdyn), f16
__global__ void k_dyn(const float* __restrict__ x, const float* __restrict__ Wd,
                      const float* __restrict__ bd, f16* __restrict__ seq) {
  long idx = (long)blockIdx.x * 256 + threadIdx.x;
  if (idx >= (long)TT * NV * DYNC) return;
  int k = (int)(idx & 63);
  long r = idx >> 6;
  int nv = (int)(r % NV);
  int t  = (int)(r / NV);
  int n = nv / VV, v = nv % VV;
  const float* xp = x + ((long)(n * TT + t) * VV + v) * 2;
  float s = bd[k] + xp[0] * Wd[k * 2] + xp[1] * Wd[k * 2 + 1];
  seq[idx] = (f16)lrelu(s);
}

// zero the decoder-input pad columns 104..127 of dfeat
__global__ void k_padfeat(f16* __restrict__ dfeat) {
  int idx = blockIdx.x * 256 + threadIdx.x;
  if (idx >= NV * 24) return;
  int nv = idx / 24, j = idx % 24;
  dfeat[(long)nv * 128 + 104 + j] = (f16)0.f;
}

// ======================= fused LSTM step (enc + dec) =======================
// Block: 16 rows x 512 gate columns. 8 waves, each wave 4 N-tiles, K in 32-chunks.
// A = [x_t | h_prev] (enc) or h_prev (dec, with precomputed Zx added).
__global__ void __launch_bounds__(256)
k_lstm_step(const f16* __restrict__ xa, int xpitch, int kcx,
            const f16* __restrict__ hprev,
            const float* __restrict__ zx,
            const float* __restrict__ bias,
            const f16* __restrict__ wpack, int kctot,
            float* __restrict__ cbuf, f16* __restrict__ hout,
            const float* __restrict__ Wout, const float* __restrict__ bout,
            float* __restrict__ outbuf, int tstep, int first) {
  __shared__ float zs[16 * 512];
  __shared__ float hS[16 * 128];
  const int tid = threadIdx.x;
  const int lane = tid & 31, wave = tid >> 5;
  const int rowBase = blockIdx.x * 16;
  const v8f vzero = {};
  v8f acc[4];
#pragma unroll
  for (int j = 0; j < 4; ++j) acc[j] = vzero;

  __builtin_prefetch(wpack + (long)wave * 4 * kctot * 512, 0, 0);

  const int kcEnd = first ? kcx : kctot;
  for (int kc = 0; kc < kcEnd; ++kc) {
    v16h a;
    if (kc < kcx) a = ldA(xa + (long)rowBase * xpitch, xpitch, kc * 32, lane);
    else          a = ldA(hprev + (long)rowBase * 128, 128, (kc - kcx) * 32, lane);
#pragma unroll
    for (int j = 0; j < 4; ++j) {
      int nt = wave * 4 + j;
      v16h b = ldB_pack(wpack + (long)(nt * kctot + kc) * 512, lane);
      acc[j] = wmma(a, b, acc[j]);
    }
  }
  // spill z tiles (+bias, +precomputed input part) to LDS
#pragma unroll
  for (int j = 0; j < 4; ++j) {
    int nt = wave * 4 + j;
    int col = nt * 16 + (lane & 15);
#pragma unroll
    for (int r = 0; r < 8; ++r) {
      int m = r + ((lane >> 4) << 3);
      float z = acc[j][r];
      if (bias) z += bias[col];
      if (zx)   z += zx[(long)(rowBase + m) * 512 + col];
      zs[m * 512 + col] = z;
    }
  }
  __syncthreads();
  // gate math: 2048 h values, 8 per thread
#pragma unroll
  for (int r = 0; r < 8; ++r) {
    int e = tid + 256 * r;
    int m = e >> 7, j = e & 127;
    int row = rowBase + m;
    float zi = zs[m * 512 + j];
    float zf = zs[m * 512 + 128 + j];
    float zg = zs[m * 512 + 256 + j];
    float zo = zs[m * 512 + 384 + j];
    float cp = first ? 0.f : cbuf[(long)row * 128 + j];
    float c  = sigm(zf) * cp + sigm(zi) * tanhf(zg);
    float h  = sigm(zo) * tanhf(c);
    cbuf[(long)row * 128 + j] = c;
    hout[(long)row * 128 + j] = (f16)h;
    if (outbuf) hS[m * 128 + j] = h;
  }
  if (outbuf) {   // decoder: fused output projection + transforms
    __syncthreads();
    if (tid < 80) {
      int m = tid / 5, jj = tid % 5;
      int row = rowBase + m;
      float s = bout[jj];
      for (int k = 0; k < 128; ++k) s += hS[m * 128 + k] * Wout[jj * 128 + k];
      if (jj == 2 || jj == 3) s = __expf(s);
      else if (jj == 4)       s = tanhf(s);
      int n = row / VV, v = row % VV;
      outbuf[(((long)n * PRED + tstep) * VV + v) * 5 + jj] = s;
    }
  }
}

// ======================= ST-GCN GEMM kernels =======================

// conv1: rows (t,n,v) over hs[t][nv][128], K=128, N=64 -> g1[n][c][t][v] f16 (raw)
__global__ void __launch_bounds__(256)
k_conv1(const f16* __restrict__ hs, const f16* __restrict__ wpack,
        const float* __restrict__ bg1, f16* __restrict__ g1) {
  int tid = threadIdx.x, lane = tid & 31, wave = tid >> 5;
  int mtile = wave >> 2, ntl = wave & 3;
  int rowBase = blockIdx.x * 32 + mtile * 16;
  v8f acc = {};
  for (int kc = 0; kc < 4; ++kc) {
    v16h a = ldA(hs + (long)rowBase * 128, 128, kc * 32, lane);
    v16h b = ldB_pack(wpack + (long)(ntl * 4 + kc) * 512, lane);
    acc = wmma(a, b, acc);
  }
  int c = ntl * 16 + (lane & 15);
#pragma unroll
  for (int r = 0; r < 8; ++r) {
    int m = r + ((lane >> 4) << 3);
    int grow = rowBase + m;
    int t = grow / NV, nv = grow % NV;
    int n = nv / VV, v = nv % VV;
    g1[(((long)n * 64 + c) * TT + t) * VV + v] = (f16)(acc[r] + bg1[c]);
  }
}

// adjacency einsum: per n, [(c,t)=512 rows] x A[n][96][96]; out y2[n][t][w][64] = lrelu(.)
__global__ void __launch_bounds__(256)
k_einsum(const f16* __restrict__ g1, const f16* __restrict__ adj, f16* __restrict__ y2) {
  int tid = threadIdx.x, lane = tid & 31, wave = tid >> 5;
  int b = blockIdx.x;
  int n = b / 24, bl = b % 24;
  int tile = bl * 8 + wave;
  int mtile = tile / 6, ntile = tile % 6;
  const f16* Ab = g1 + (long)n * 64 * TT * VV + (long)mtile * 16 * VV;
  const f16* Bb = adj + (long)n * VV * VV;
  v8f acc = {};
  for (int kc = 0; kc < 3; ++kc) {
    v16h a  = ldA(Ab, VV, kc * 32, lane);
    v16h bb = ldB_dir(Bb, VV, kc * 32, ntile * 16, lane);
    acc = wmma(a, bb, acc);
  }
  int w = ntile * 16 + (lane & 15);
#pragma unroll
  for (int r = 0; r < 8; ++r) {
    int mrow = mtile * 16 + r + ((lane >> 4) << 3);
    int c = mrow >> 3, t = mrow & 7;
    y2[(((long)n * TT + t) * VV + w) * 64 + c] = (f16)lrelu(acc[r]);
  }
}

// tcn1: rows (n,t,w), K=64, N=64; t1 = lrelu(y2 @ Wt1^T + bt1)  (y2 already = lrelu(einsum))
__global__ void __launch_bounds__(256)
k_tcn1(const f16* __restrict__ y2, const f16* __restrict__ wpack,
       const float* __restrict__ bt1, f16* __restrict__ t1) {
  int tid = threadIdx.x, lane = tid & 31, wave = tid >> 5;
  int mtile = wave >> 2, ntl = wave & 3;
  int rowBase = blockIdx.x * 32 + mtile * 16;
  v8f acc = {};
  for (int kc = 0; kc < 2; ++kc) {
    v16h a = ldA(y2 + (long)rowBase * 64, 64, kc * 32, lane);
    v16h b = ldB_pack(wpack + (long)(ntl * 2 + kc) * 512, lane);
    acc = wmma(a, b, acc);
  }
  int col = ntl * 16 + (lane & 15);
#pragma unroll
  for (int r = 0; r < 8; ++r) {
    int m = r + ((lane >> 4) << 3);
    t1[(long)(rowBase + m) * 64 + col] = (f16)lrelu(acc[r] + bt1[col]);
  }
}

// residual conv path (layers 2/3): rbuf = A @ Wr^T + br, plus BN sum/sumsq stats
__global__ void __launch_bounds__(256)
k_res16(const f16* __restrict__ abuf, int apitch, int klim, int kcN,
        const f16* __restrict__ wpack, const float* __restrict__ br,
        int cout, float* __restrict__ rbuf, float* __restrict__ gstats) {
  __shared__ float ssum[16], ssq[16];
  int tid = threadIdx.x, lane = tid & 31, wave = tid >> 5;
  if (tid < 16) { ssum[tid] = 0.f; ssq[tid] = 0.f; }
  __syncthreads();
  int rowBase = blockIdx.x * 128 + wave * 16;
  v8f acc = {};
  for (int kc = 0; kc < kcN; ++kc) {
    v16h a = ldA_g<false>(abuf + (long)rowBase * apitch, apitch, kc * 32, klim, lane);
    v16h b = ldB_pack(wpack + (long)kc * 512, lane);
    acc = wmma(a, b, acc);
  }
  int col = lane & 15;
  if (col < cout) {
    float ls = 0.f, lq = 0.f;
#pragma unroll
    for (int r = 0; r < 8; ++r) {
      int m = r + ((lane >> 4) << 3);
      float val = acc[r] + br[col];
      rbuf[(long)(rowBase + m) * 16 + col] = val;
      ls += val; lq += val * val;
    }
    atomicAdd(&ssum[col], ls);
    atomicAdd(&ssq[col], lq);
  }
  __syncthreads();
  if (tid < cout) {
    atomicAdd(&gstats[tid], ssum[tid]);
    atomicAdd(&gstats[16 + tid], ssq[tid]);
  }
}

// BN finalize: sc[c] = gamma*rstd ; sc[16+c] = beta - mean*gamma*rstd
__global__ void k_bnfin(const float* __restrict__ gstats, const float* __restrict__ g,
                        const float* __restrict__ be, float* __restrict__ sc, int cout) {
  int t = threadIdx.x;
  if (t < cout) {
    float mean = gstats[t] / (float)NTV;
    float var  = gstats[16 + t] / (float)NTV - mean * mean;
    float rstd = rsqrtf(var + 1e-5f);
    float scale = g[t] * rstd;
    sc[t] = scale;
    sc[16 + t] = be[t] - mean * scale;
  }
}

// tcn (layers 2/3): out = lrelu( lrelu(A) @ Wt^T + bt + BN(rbuf) )
__global__ void __launch_bounds__(256)
k_tcn16(const f16* __restrict__ abuf, int apitch, int klim, int kcN,
        const f16* __restrict__ wpack, const float* __restrict__ bt,
        const float* __restrict__ rbuf, const float* __restrict__ sc,
        int cout, f16* __restrict__ outH, float* __restrict__ outF) {
  int tid = threadIdx.x, lane = tid & 31, wave = tid >> 5;
  int rowBase = blockIdx.x * 128 + wave * 16;
  v8f acc = {};
  for (int kc = 0; kc < kcN; ++kc) {
    v16h a = ldA_g<true>(abuf + (long)rowBase * apitch, apitch, kc * 32, klim, lane);
    v16h b = ldB_pack(wpack + (long)kc * 512, lane);
    acc = wmma(a, b, acc);
  }
  int col = lane & 15;
  if (col < cout) {
#pragma unroll
    for (int r = 0; r < 8; ++r) {
      int m = r + ((lane >> 4) << 3);
      long row = rowBase + m;
      float val = acc[r] + bt[col] + rbuf[row * 16 + col] * sc[col] + sc[16 + col];
      float y = lrelu(val);
      if (outH) outH[row * 16 + col] = (f16)y;
      else      outF[row * 8 + col]  = y;
    }
  }
}

// MaxPool2d((1,2), pad (0,1)) over T=8 -> 5, into dfeat cols [0..39] (f16)
__global__ void k_pool(const float* __restrict__ y3, f16* __restrict__ dfeat) {
  int idx = blockIdx.x * 256 + threadIdx.x;
  if (idx >= NV * 8) return;
  int nv = idx >> 3, c = idx & 7;
  int n = nv / VV, v = nv % VV;
  const float* p = y3 + ((long)(n * TT) * VV + v) * 8 + c;  // stride per t = VV*8
  float o0 = p[0];
  float o1 = fmaxf(p[1 * VV * 8], p[2 * VV * 8]);
  float o2 = fmaxf(p[3 * VV * 8], p[4 * VV * 8]);
  float o3 = fmaxf(p[5 * VV * 8], p[6 * VV * 8]);
  float o4 = p[7 * VV * 8];
  f16* d = dfeat + (long)nv * 128 + c * 5;
  d[0] = (f16)o0; d[1] = (f16)o1; d[2] = (f16)o2; d[3] = (f16)o3; d[4] = (f16)o4;
}

// hsel: lrelu(hT @ Whid^T + bhid) -> dfeat cols [40..103]
__global__ void __launch_bounds__(256)
k_hsel(const f16* __restrict__ hT, const f16* __restrict__ wpack,
       const float* __restrict__ bhid, f16* __restrict__ dfeat) {
  int tid = threadIdx.x, lane = tid & 31, wave = tid >> 5;
  int mtile = wave >> 2, ntl = wave & 3;
  int rowBase = blockIdx.x * 32 + mtile * 16;
  v8f acc = {};
  for (int kc = 0; kc < 4; ++kc) {
    v16h a = ldA(hT + (long)rowBase * 128, 128, kc * 32, lane);
    v16h b = ldB_pack(wpack + (long)(ntl * 4 + kc) * 512, lane);
    acc = wmma(a, b, acc);
  }
  int col = ntl * 16 + (lane & 15);
#pragma unroll
  for (int r = 0; r < 8; ++r) {
    int m = r + ((lane >> 4) << 3);
    dfeat[(long)(rowBase + m) * 128 + 40 + col] = (f16)lrelu(acc[r] + bhid[col]);
  }
}

// Zx = dfeat @ Wih_d^T + (bih_d + bhh_d) : rows NV, K=128(pad), N=512, f32
__global__ void __launch_bounds__(256)
k_zxd(const f16* __restrict__ dfeat, const f16* __restrict__ wpack,
      const float* __restrict__ biasd, float* __restrict__ zx) {
  int tid = threadIdx.x, lane = tid & 31, wave = tid >> 5;
  int rowBase = blockIdx.x * 16;
  const v8f vzero = {};
  v8f acc[4];
#pragma unroll
  for (int j = 0; j < 4; ++j) acc[j] = vzero;
  for (int kc = 0; kc < 4; ++kc) {
    v16h a = ldA(dfeat + (long)rowBase * 128, 128, kc * 32, lane);
#pragma unroll
    for (int j = 0; j < 4; ++j) {
      int nt = wave * 4 + j;
      v16h b = ldB_pack(wpack + (long)(nt * 4 + kc) * 512, lane);
      acc[j] = wmma(a, b, acc[j]);
    }
  }
#pragma unroll
  for (int j = 0; j < 4; ++j) {
    int nt = wave * 4 + j;
    int col = nt * 16 + (lane & 15);
#pragma unroll
    for (int r = 0; r < 8; ++r) {
      int m = r + ((lane >> 4) << 3);
      zx[(long)(rowBase + m) * 512 + col] = acc[j][r] + biasd[col];
    }
  }
}

// ======================= host launcher =======================

extern "C" void kernel_launch(void* const* d_in, const int* in_sizes, int n_in,
                              void* d_out, int out_size, void* d_ws, size_t ws_size,
                              hipStream_t stream) {
  const float* x     = (const float*)d_in[0];
  const float* Aadj  = (const float*)d_in[1];
  const float* Wdyn  = (const float*)d_in[2];
  const float* bdyn  = (const float*)d_in[3];
  const float* Wih_e = (const float*)d_in[4];
  const float* Whh_e = (const float*)d_in[5];
  const float* bih_e = (const float*)d_in[6];
  const float* bhh_e = (const float*)d_in[7];
  const float* Wg1   = (const float*)d_in[8];
  const float* bg1   = (const float*)d_in[9];
  const float* Wt1   = (const float*)d_in[10];
  const float* bt1   = (const float*)d_in[11];
  const float* Wr2   = (const float*)d_in[12];
  const float* br2   = (const float*)d_in[13];
  const float* g2    = (const float*)d_in[14];
  const float* be2   = (const float*)d_in[15];
  const float* Wt2   = (const float*)d_in[16];
  const float* bt2   = (const float*)d_in[17];
  const float* Wr3   = (const float*)d_in[18];
  const float* br3   = (const float*)d_in[19];
  const float* g3    = (const float*)d_in[20];
  const float* be3   = (const float*)d_in[21];
  const float* Wt3   = (const float*)d_in[22];
  const float* bt3   = (const float*)d_in[23];
  const float* Whid  = (const float*)d_in[24];
  const float* bhid  = (const float*)d_in[25];
  const float* Wih_d = (const float*)d_in[26];
  const float* Whh_d = (const float*)d_in[27];
  const float* bih_d = (const float*)d_in[28];
  const float* bhh_d = (const float*)d_in[29];
  const float* Wout  = (const float*)d_in[30];
  const float* bout  = (const float*)d_in[31];
  float* out = (float*)d_out;

  char* ws = (char*)d_ws;
  size_t off = 0;
  auto alloc = [&](size_t bytes) -> void* {
    off = (off + 255) & ~(size_t)255;
    void* p = ws + off;
    off += bytes;
    return p;
  };

  // ---- static region (zeroed every call): packed weights + stats + bias/scales
  f16* we_p    = (f16*)alloc((size_t)32 * 6 * 512 * 2);  // enc [Wih|Whh], K=192
  f16* wdhh_p  = (f16*)alloc((size_t)32 * 4 * 512 * 2);  // dec Whh, K=128
  f16* wdih_p  = (f16*)alloc((size_t)32 * 4 * 512 * 2);  // dec Wih, K=104 pad 128
  f16* whid_p  = (f16*)alloc((size_t)4  * 4 * 512 * 2);
  f16* wg1_p   = (f16*)alloc((size_t)4  * 4 * 512 * 2);
  f16* wt1_p   = (f16*)alloc((size_t)4  * 2 * 512 * 2);
  f16* wr2_p   = (f16*)alloc((size_t)1  * 2 * 512 * 2);
  f16* wt2_p   = (f16*)alloc((size_t)1  * 2 * 512 * 2);
  f16* wr3_p   = (f16*)alloc((size_t)1  * 1 * 512 * 2);
  f16* wt3_p   = (f16*)alloc((size_t)1  * 1 * 512 * 2);
  float* gst2  = (float*)alloc(32 * 4);
  float* gst3  = (float*)alloc(32 * 4);
  float* sc2   = (float*)alloc(32 * 4);
  float* sc3   = (float*)alloc(32 * 4);
  float* biasE = (float*)alloc(512 * 4);
  float* biasD = (float*)alloc(512 * 4);
  size_t staticEnd = (off + 255) & ~(size_t)255;

  // ---- big buffers (with lifetime aliasing)
  f16*  adjh  = (f16*)alloc((size_t)NB * VV * VV * 2);          // 9.4 MB
  f16*  seq   = (f16*)alloc((size_t)TT * NV * 64 * 2);          // 50.3 MB (reused: y2)
  f16*  hs    = (f16*)alloc((size_t)TT * NV * 128 * 2);         // 100.7 MB (reused: Zx f32, same bytes)
  float* cbuf = (float*)alloc((size_t)NV * 128 * 4);            // 25.2 MB (enc c, reused dec c)
  f16*  g1    = (f16*)alloc((size_t)NB * 64 * TT * VV * 2);     // 50.3 MB (reused: y3 f32)
  f16*  t1    = (f16*)alloc((size_t)NTV * 64 * 2);              // 50.3 MB (reused: h_dec)
  float* rbuf = (float*)alloc((size_t)NTV * 16 * 4);            // 25.2 MB
  f16*  y2b   = (f16*)alloc((size_t)NTV * 16 * 2);              // 12.6 MB
  f16*  dfeat = (f16*)alloc((size_t)NV * 128 * 2);              // 12.6 MB
  f16*  y2    = seq;
  float* zxd  = (float*)hs;
  float* y3   = (float*)g1;
  f16*  hdec  = t1;
  (void)ws_size; (void)in_sizes; (void)n_in; (void)out_size;

  // ---- 0) zero static region (pack pads, BN stats)
  k_zero<<<2048, 256, 0, stream>>>((unsigned int*)ws, (long)(staticEnd / 4));

  // ---- 1) weight packing / conversions
  auto packGrid = [](int n, int k) { return (n * k + 255) / 256; };
  k_pack_w<<<packGrid(512, 64), 256, 0, stream>>>(Wih_e, we_p, 512, 64, 6, 0);
  k_pack_w<<<packGrid(512, 128), 256, 0, stream>>>(Whh_e, we_p, 512, 128, 6, 64);
  k_pack_w<<<packGrid(512, 128), 256, 0, stream>>>(Whh_d, wdhh_p, 512, 128, 4, 0);
  k_pack_w<<<packGrid(512, 104), 256, 0, stream>>>(Wih_d, wdih_p, 512, 104, 4, 0);
  k_pack_w<<<packGrid(64, 128), 256, 0, stream>>>(Whid, whid_p, 64, 128, 4, 0);
  k_pack_w<<<packGrid(64, 128), 256, 0, stream>>>(Wg1, wg1_p, 64, 128, 4, 0);
  k_pack_w<<<packGrid(64, 64), 256, 0, stream>>>(Wt1, wt1_p, 64, 64, 2, 0);
  k_pack_w<<<packGrid(16, 64), 256, 0, stream>>>(Wr2, wr2_p, 16, 64, 2, 0);
  k_pack_w<<<packGrid(16, 64), 256, 0, stream>>>(Wt2, wt2_p, 16, 64, 2, 0);
  k_pack_w<<<packGrid(8, 16), 256, 0, stream>>>(Wr3, wr3_p, 8, 16, 1, 0);
  k_pack_w<<<packGrid(8, 16), 256, 0, stream>>>(Wt3, wt3_p, 8, 16, 1, 0);
  k_cvt<<<(NB * VV * VV + 255) / 256, 256, 0, stream>>>(Aadj, adjh, (long)NB * VV * VV);
  k_addb<<<2, 256, 0, stream>>>(bih_e, bhh_e, biasE, 512);
  k_addb<<<2, 256, 0, stream>>>(bih_d, bhh_d, biasD, 512);
  k_padfeat<<<(NV * 24 + 255) / 256, 256, 0, stream>>>(dfeat);

  // ---- 2) dyn projection
  k_dyn<<<(TT * NV * DYNC + 255) / 256, 256, 0, stream>>>(x, Wdyn, bdyn, seq);

  // ---- 3) encoder LSTM, 8 steps (kernel boundary = global sync)
  for (int t = 0; t < TT; ++t) {
    k_lstm_step<<<NV / 16, 256, 0, stream>>>(
        seq + (size_t)t * NV * 64, 64, 2,
        hs + (size_t)(t > 0 ? t - 1 : 0) * NV * 128,
        nullptr, biasE, we_p, 6,
        cbuf, hs + (size_t)t * NV * 128,
        nullptr, nullptr, nullptr, 0, t == 0 ? 1 : 0);
  }

  // ---- 4) ST-GCN layer 1
  k_conv1<<<NTV / 32, 256, 0, stream>>>(hs, wg1_p, bg1, g1);
  k_einsum<<<NB * 24, 256, 0, stream>>>(g1, adjh, y2);
  k_tcn1<<<NTV / 32, 256, 0, stream>>>(y2, wt1_p, bt1, t1);

  // ---- 5) layer 2 (residual conv + BN + tcn)
  k_res16<<<NTV / 128, 256, 0, stream>>>(t1, 64, 64, 2, wr2_p, br2, 16, rbuf, gst2);
  k_bnfin<<<1, 32, 0, stream>>>(gst2, g2, be2, sc2, 16);
  k_tcn16<<<NTV / 128, 256, 0, stream>>>(t1, 64, 64, 2, wt2_p, bt2, rbuf, sc2, 16, y2b, nullptr);

  // ---- 6) layer 3
  k_res16<<<NTV / 128, 256, 0, stream>>>(y2b, 16, 16, 1, wr3_p, br3, 8, rbuf, gst3);
  k_bnfin<<<1, 32, 0, stream>>>(gst3, g3, be3, sc3, 8);
  k_tcn16<<<NTV / 128, 256, 0, stream>>>(y2b, 16, 16, 1, wt3_p, bt3, rbuf, sc3, 8, nullptr, y3);

  // ---- 7) pooling + hidden selector -> decoder features
  k_pool<<<(NV * 8 + 255) / 256, 256, 0, stream>>>(y3, dfeat);
  k_hsel<<<NV / 32, 256, 0, stream>>>(hs + (size_t)7 * NV * 128, whid_p, bhid, dfeat);

  // ---- 8) decoder input GEMM once (input constant over steps)
  k_zxd<<<NV / 16, 256, 0, stream>>>(dfeat, wdih_p, biasD, zxd);

  // ---- 9) decoder LSTM, 12 steps, fused output head
  for (int t = 0; t < PRED; ++t) {
    k_lstm_step<<<NV / 16, 256, 0, stream>>>(
        nullptr, 0, 0, hdec, zxd, nullptr, wdhh_p, 4,
        cbuf, hdec, Wout, bout, out, t, t == 0 ? 1 : 0);
  }
}
